// SorscherRNN_47510928228535
// MI455X (gfx1250) — matrix operand order repaired and checked by reading, not compiled, not run on hardware
//
#include <hip/hip_runtime.h>
#include <hip/hip_bf16.h>

// Sorscher RNN: Ng=4096, Np=512, B=128, T=100
#define NG 4096
#define NP 512
#define BB 128
#define TT 100

#define KC   32   // K-chunk per WMMA
#define APAD 40   // padded LDS row stride (elements) to spread bank usage

typedef __attribute__((ext_vector_type(16))) __bf16 v16bf;
typedef __attribute__((ext_vector_type(8)))  float v8f;

union FragU {
    v16bf v;
    uint4 q[2];
};

typedef __attribute__((address_space(3))) unsigned short lds_us_t;

static __device__ __forceinline__ unsigned lds_addr(const unsigned short* p) {
    // generic (known-LDS) pointer -> 32-bit LDS byte offset
    return (unsigned)(unsigned long long)(lds_us_t*)p;
}

static __device__ __forceinline__ void async_copy_b128(const unsigned short* g, unsigned loff) {
    // GLOBAL_LOAD_ASYNC_TO_LDS_B128: LDS[loff] = MEM[g..g+15], tracked by ASYNCcnt
    asm volatile("global_load_async_to_lds_b128 %0, %1, off" :: "v"(loff), "v"(g) : "memory");
}

static __device__ __forceinline__ unsigned short f32_to_bf16(float f) {
    unsigned int u = __float_as_uint(f);
    unsigned int rnd = 0x7FFFu + ((u >> 16) & 1u);
    u += rnd;
    return (unsigned short)(u >> 16);
}

// A fragment (16x32 bf16, MxK) from LDS chunk (row-major, stride APAD):
// lane L -> m = L%16, kh = L/16; elems 0..7: k=8*kh+e ; elems 8..15: k=16+8*kh+(e-8)
static __device__ __forceinline__ v16bf lds_a_frag(const unsigned short* S, int row0, int lane) {
    int m  = lane & 15;
    int kh = lane >> 4;
    const unsigned short* p = S + (row0 + m) * APAD + 8 * kh;
    FragU f;
    f.q[0] = *(const uint4*)(p);
    f.q[1] = *(const uint4*)(p + 16);
    return f.v;
}

// B fragment (32x16 bf16, KxN) from LDS chunk of Bm rows (N-major, stride APAD):
// lane L -> n = L%16, kh = L/16; elems 0..7: k=8*kh+e ; elems 8..15: k=16+8*kh+(e-8)
static __device__ __forceinline__ v16bf lds_b_frag(const unsigned short* S, int col0, int lane) {
    int n  = lane & 15;
    int kh = lane >> 4;
    const unsigned short* p = S + (col0 + n) * APAD + 8 * kh;
    FragU f;
    f.q[0] = *(const uint4*)(p);
    f.q[1] = *(const uint4*)(p + 16);
    return f.v;
}

static __device__ __forceinline__ v8f wmma_bf16(v16bf a, v16bf b, v8f c) {
    return __builtin_amdgcn_wmma_f32_16x16x32_bf16(false, a, false, b, (short)0, c, false, false);
}

// Cooperatively (256 threads) async-copy one K-chunk:
//  A tile 128x32 (2 b128 pieces per thread), B tile 64x32 (1 piece per thread).
// => 3 async ops per wave per chunk.
static __device__ __forceinline__ void copy_chunk(const unsigned short* __restrict__ A,
                                                  const unsigned short* __restrict__ Bm,
                                                  int colBase, int k0, int K,
                                                  unsigned short* As, unsigned short* Bs,
                                                  int tid) {
#pragma unroll
    for (int j = 0; j < 2; ++j) {
        int p = tid * 2 + j;
        int r = p >> 2, s = p & 3;
        async_copy_b128(A + (size_t)r * K + k0 + s * 8, lds_addr(As + r * APAD + s * 8));
    }
    {
        int r = tid >> 2, s = tid & 3;
        async_copy_b128(Bm + (size_t)(colBase + r) * K + k0 + s * 8,
                        lds_addr(Bs + r * APAD + s * 8));
    }
}

// Triple-buffered async-staged GEMM core: C(128x64) += A(128xK) @ Bm(colBase..+64, K)^T
// 8 waves: waveM = wave&3 (rows), waveN = wave>>2 (cols); 32x32 per wave (2x2 WMMA).
template <int KTOT>
static __device__ __forceinline__ void gemm_core_128x64(const unsigned short* __restrict__ A,
                                                        const unsigned short* __restrict__ Bm,
                                                        int colBase,
                                                        unsigned short* As,  // [3][128*APAD]
                                                        unsigned short* Bs,  // [3][64*APAD]
                                                        v8f acc[2][2]) {
    const int tid  = threadIdx.x;
    const int lane = tid & 31;
    const int wave = tid >> 5;
    const int row0 = (wave & 3) * 32;
    const int coll = (wave >> 2) * 32;
    const int NKC  = KTOT / KC;

    copy_chunk(A, Bm, colBase, 0, KTOT, As, Bs, tid);
    if (NKC > 1) copy_chunk(A, Bm, colBase, KC, KTOT, As + 128 * APAD, Bs + 64 * APAD, tid);

    for (int kc = 0; kc < NKC; ++kc) {
        // ensure chunk kc has landed (async loads complete in order; allow the
        // newest in-flight chunk's 3 ops to remain outstanding)
        if (kc + 1 < NKC) asm volatile("s_wait_asynccnt 0x3" ::: "memory");
        else              asm volatile("s_wait_asynccnt 0x0" ::: "memory");
        __syncthreads();

        const unsigned short* Ac = As + (kc % 3) * (128 * APAD);
        const unsigned short* Bc = Bs + (kc % 3) * (64 * APAD);
        v16bf a0 = lds_a_frag(Ac, row0,      lane);
        v16bf a1 = lds_a_frag(Ac, row0 + 16, lane);
        v16bf b0 = lds_b_frag(Bc, coll,      lane);
        v16bf b1 = lds_b_frag(Bc, coll + 16, lane);
        acc[0][0] = wmma_bf16(a0, b0, acc[0][0]);
        acc[0][1] = wmma_bf16(a0, b1, acc[0][1]);
        acc[1][0] = wmma_bf16(a1, b0, acc[1][0]);
        acc[1][1] = wmma_bf16(a1, b1, acc[1][1]);

        __syncthreads();  // everyone done reading buf kc%3 before it is overwritten
        if (kc + 2 < NKC)
            copy_chunk(A, Bm, colBase, (kc + 2) * KC, KTOT,
                       As + ((kc + 2) % 3) * (128 * APAD),
                       Bs + ((kc + 2) % 3) * (64 * APAD), tid);
    }
}

// ---------------------------------------------------------------------------
// fp32 -> bf16 conversion
// ---------------------------------------------------------------------------
__global__ __launch_bounds__(256) void cvt_bf16_kernel(const float* __restrict__ in,
                                                       unsigned short* __restrict__ out, int n) {
    int i = blockIdx.x * 256 + threadIdx.x;
    if (i < n) out[i] = f32_to_bf16(in[i]);
}

// ---------------------------------------------------------------------------
// Encoder: h0 = p0 @ W_enc^T.  M=128, N=4096, K=512.  grid.x = NG/64.
// ---------------------------------------------------------------------------
__global__ __launch_bounds__(256) void encoder_kernel(const unsigned short* __restrict__ p0b,
                                                      const unsigned short* __restrict__ Wencb,
                                                      unsigned short* __restrict__ h0) {
    __shared__ unsigned short As[3 * 128 * APAD];
    __shared__ unsigned short Bs[3 * 64 * APAD];
    int lane = threadIdx.x & 31;
    int wave = threadIdx.x >> 5;
    int row0 = (wave & 3) * 32;
    int col0 = blockIdx.x * 64 + (wave >> 2) * 32;

    v8f acc[2][2] = {};
    gemm_core_128x64<NP>(p0b, Wencb, blockIdx.x * 64, As, Bs, acc);

    int nl = lane & 15, kh = lane >> 4;
    for (int mt = 0; mt < 2; ++mt)
        for (int nt = 0; nt < 2; ++nt)
            for (int r = 0; r < 8; ++r) {
                int b = row0 + mt * 16 + r + 8 * kh;
                int g = col0 + nt * 16 + nl;
                h0[(size_t)b * NG + g] = f32_to_bf16(acc[mt][nt][r]);
            }
}

// ---------------------------------------------------------------------------
// RNN step: hout = relu(vin_t + hprev @ W_hh^T).  M=128, N=4096, K=4096.
// grid.x = NG/64.
// ---------------------------------------------------------------------------
__global__ __launch_bounds__(256) void rnn_step_kernel(const unsigned short* __restrict__ hprev,
                                                       const unsigned short* __restrict__ Whhb,
                                                       const float* __restrict__ v,
                                                       const float* __restrict__ Wih,
                                                       unsigned short* __restrict__ hout,
                                                       int t) {
    __shared__ unsigned short As[3 * 128 * APAD];
    __shared__ unsigned short Bs[3 * 64 * APAD];
    int lane = threadIdx.x & 31;
    int wave = threadIdx.x >> 5;
    int row0 = (wave & 3) * 32;
    int col0 = blockIdx.x * 64 + (wave >> 2) * 32;

    v8f acc[2][2] = {};
    gemm_core_128x64<NG>(hprev, Whhb, blockIdx.x * 64, As, Bs, acc);

    int nl = lane & 15, kh = lane >> 4;
    for (int nt = 0; nt < 2; ++nt) {
        int g = col0 + nt * 16 + nl;
        float w0 = Wih[g * 2 + 0];
        float w1 = Wih[g * 2 + 1];
        for (int mt = 0; mt < 2; ++mt)
            for (int r = 0; r < 8; ++r) {
                int b = row0 + mt * 16 + r + 8 * kh;
                float vin = v[((size_t)b * TT + t) * 2 + 0] * w0 +
                            v[((size_t)b * TT + t) * 2 + 1] * w1;
                float val = acc[mt][nt][r] + vin;
                val = val > 0.0f ? val : 0.0f;
                hout[(size_t)b * NG + g] = f32_to_bf16(val);
            }
    }
}

// ---------------------------------------------------------------------------
// Decoder: logits[b,t,:] = hs[t,b,:] @ W_dec^T.  grid = (NP/64, T).
// Output f32 (pre-softmax) permuted to (B,T,Np).
// ---------------------------------------------------------------------------
__global__ __launch_bounds__(256) void decoder_kernel(const unsigned short* __restrict__ hs,
                                                      const unsigned short* __restrict__ Wdecb,
                                                      float* __restrict__ out) {
    __shared__ unsigned short As[3 * 128 * APAD];
    __shared__ unsigned short Bs[3 * 64 * APAD];
    int lane = threadIdx.x & 31;
    int wave = threadIdx.x >> 5;
    int t = blockIdx.y;
    int row0 = (wave & 3) * 32;
    int col0 = blockIdx.x * 64 + (wave >> 2) * 32;
    const unsigned short* A = hs + (size_t)t * BB * NG;

    v8f acc[2][2] = {};
    gemm_core_128x64<NG>(A, Wdecb, blockIdx.x * 64, As, Bs, acc);

    int nl = lane & 15, kh = lane >> 4;
    for (int mt = 0; mt < 2; ++mt)
        for (int nt = 0; nt < 2; ++nt)
            for (int r = 0; r < 8; ++r) {
                int b = row0 + mt * 16 + r + 8 * kh;
                int n = col0 + nt * 16 + nl;
                out[((size_t)b * TT + t) * NP + n] = acc[mt][nt][r];
            }
}

// ---------------------------------------------------------------------------
// In-place log_softmax over rows of 512 f32.  One wave per row, 16 elems/lane.
// ---------------------------------------------------------------------------
__global__ __launch_bounds__(256) void log_softmax_kernel(float* __restrict__ out) {
    int lane = threadIdx.x & 31;
    int wave = threadIdx.x >> 5;
    size_t row = (size_t)blockIdx.x * 8 + wave;
    float* p = out + row * NP;

    float x[16];
    float mx = -3.402823466e38f;
    for (int i = 0; i < 16; ++i) {
        x[i] = p[lane + i * 32];
        mx = fmaxf(mx, x[i]);
    }
    for (int off = 16; off > 0; off >>= 1) mx = fmaxf(mx, __shfl_xor(mx, off, 32));
    float s = 0.0f;
    for (int i = 0; i < 16; ++i) s += expf(x[i] - mx);
    for (int off = 16; off > 0; off >>= 1) s += __shfl_xor(s, off, 32);
    float lse = logf(s) + mx;
    for (int i = 0; i < 16; ++i) p[lane + i * 32] = x[i] - lse;
}

// ---------------------------------------------------------------------------
extern "C" void kernel_launch(void* const* d_in, const int* in_sizes, int n_in,
                              void* d_out, int out_size, void* d_ws, size_t ws_size,
                              hipStream_t stream) {
    const float* v     = (const float*)d_in[0]; // (B,T,2)
    const float* p0    = (const float*)d_in[1]; // (B,Np)
    const float* W_enc = (const float*)d_in[2]; // (Ng,Np)
    const float* W_ih  = (const float*)d_in[3]; // (Ng,2)
    const float* W_hh  = (const float*)d_in[4]; // (Ng,Ng)
    const float* W_dec = (const float*)d_in[5]; // (Np,Ng)
    float* out = (float*)d_out;                 // (B,T,Np) f32

    // workspace layout (bf16 copies + hidden-state history)
    char* ws = (char*)d_ws;
    unsigned short* whh_b  = (unsigned short*)ws; ws += (size_t)NG * NG * 2;  // 32 MB
    unsigned short* wenc_b = (unsigned short*)ws; ws += (size_t)NG * NP * 2;  // 4 MB
    unsigned short* wdec_b = (unsigned short*)ws; ws += (size_t)NP * NG * 2;  // 4 MB
    unsigned short* p0_b   = (unsigned short*)ws; ws += (size_t)BB * NP * 2;  // 128 KB
    unsigned short* h0_b   = (unsigned short*)ws; ws += (size_t)BB * NG * 2;  // 1 MB
    unsigned short* hs     = (unsigned short*)ws;                             // 100 MB (T,B,Ng)

    // fp32 -> bf16 weight/input conversions
    {
        int n;
        n = NG * NG; cvt_bf16_kernel<<<dim3((n + 255) / 256), 256, 0, stream>>>(W_hh, whh_b, n);
        n = NG * NP; cvt_bf16_kernel<<<dim3((n + 255) / 256), 256, 0, stream>>>(W_enc, wenc_b, n);
        n = NP * NG; cvt_bf16_kernel<<<dim3((n + 255) / 256), 256, 0, stream>>>(W_dec, wdec_b, n);
        n = BB * NP; cvt_bf16_kernel<<<dim3((n + 255) / 256), 256, 0, stream>>>(p0, p0_b, n);
    }

    // h0 = p0 @ W_enc^T
    encoder_kernel<<<dim3(NG / 64), 256, 0, stream>>>(p0_b, wenc_b, h0_b);

    // sequential RNN: one launch per step gives the required grid-wide sync
    const unsigned short* hprev = h0_b;
    for (int t = 0; t < TT; ++t) {
        unsigned short* hout = hs + (size_t)t * BB * NG;
        rnn_step_kernel<<<dim3(NG / 64), 256, 0, stream>>>(hprev, whh_b, v, W_ih, hout, t);
        hprev = hout;
    }

    // logits = hs @ W_dec^T  (permuted store to (B,T,Np))
    decoder_kernel<<<dim3(NP / 64, TT), 256, 0, stream>>>(hs, wdec_b, out);

    // in-place log_softmax over last axis (B*T = 12800 rows of 512)
    log_softmax_kernel<<<dim3(BB * TT / 8), 256, 0, stream>>>(out);
}
